// TemporalFusion_81630148428322
// MI455X (gfx1250) — matrix operand order repaired and checked by reading, not compiled, not run on hardware
//
#include <hip/hip_runtime.h>

// ---------------------------------------------------------------------------
// TemporalFusion for MI455X (gfx1250, wave32, WMMA + TDM).
//  - All matmuls on V_WMMA_F32_16X16X32_BF16 (bf16 in, f32 accum).
//  - Weight tiles staged into LDS by the Tensor Data Mover (TENSORcnt),
//    double buffered (TDM of stage s+1 overlaps WMMA of stage s);
//    B fragments read via ds_load_b128 from 16B-padded rows.
//  - ~112 GFLOP vs ~330 MB HBM traffic (14 us floor at 23.3 TB/s).
// ---------------------------------------------------------------------------

#define NTOT   90112     // B*NPG nodes
#define BGR    4096      // graphs
#define NPG    22
#define DBB    256       // backbone / hidden
#define DNODE  128
#define DPROJ  256
#define DGIN   64
#define DGOUT  128
#define DIN    512       // DBB + DPROJ
#define DFUSE  384       // DBB + DGOUT

typedef __attribute__((ext_vector_type(16))) __bf16 bf16x16;
typedef __attribute__((ext_vector_type(8)))  __bf16 bf16x8;
typedef __attribute__((ext_vector_type(8)))  float  f32x8;
typedef __attribute__((ext_vector_type(4)))  float  f32x4;
typedef __attribute__((ext_vector_type(4)))  unsigned int u32x4;
typedef __attribute__((ext_vector_type(8)))  int    i32x8;
typedef __attribute__((ext_vector_type(4)))  int    i32x4;

// ---- gfx1250 Tensor Data Mover availability -------------------------------
#if defined(__AMDGCN__) && __has_builtin(__builtin_amdgcn_tensor_load_to_lds) && \
    __has_builtin(__builtin_amdgcn_s_wait_tensorcnt)
  #define USE_TDM 1
  #if __has_include(<hip/amd_detail/amd_gfx1250_TDM.h>)
    #define TDM_ARITY 6      // amdgpu-toolchain (therock headers)
  #else
    #define TDM_ARITY 5      // ROCm 7.2 clang-22
  #endif
#else
  #define USE_TDM 0
#endif

// ---- WMMA wrapper ----------------------------------------------------------
__device__ __forceinline__ f32x8 wmma_bf16(bf16x16 a, bf16x16 b, f32x8 c) {
  return __builtin_amdgcn_wmma_f32_16x16x32_bf16(false, a, false, b,
                                                 (short)0, c, false, false);
}

// ---- A fragment (16x32 bf16), ISA layout: M=lane%16,
//      V0..3: K = 8*(lane/16)+0..7 ; V4..7: K = 16+8*(lane/16)+0..7 ----------
template <int LDA>
__device__ __forceinline__ bf16x16
fragA_f32(const float* __restrict__ base, int row, int k0, int lane) {
  const int r  = row + (lane & 15);
  const int kh = (lane & 16) ? 8 : 0;
  const float* p = base + (long)r * LDA + k0 + kh;
  f32x4 a0 = *(const f32x4*)(p);
  f32x4 a1 = *(const f32x4*)(p + 4);
  f32x4 a2 = *(const f32x4*)(p + 16);
  f32x4 a3 = *(const f32x4*)(p + 20);
  bf16x16 o;
#pragma unroll
  for (int i = 0; i < 4; ++i) {
    o[i]      = (__bf16)a0[i];
    o[4 + i]  = (__bf16)a1[i];
    o[8 + i]  = (__bf16)a2[i];
    o[12 + i] = (__bf16)a3[i];
  }
  return o;
}

template <int LDA>
__device__ __forceinline__ bf16x16
fragA_bf16(const __bf16* __restrict__ base, int row, int k0, int lane) {
  const int r  = row + (lane & 15);
  const int kh = (lane & 16) ? 8 : 0;
  const __bf16* p = base + (long)r * LDA + k0 + kh;
  bf16x8 lo = *(const bf16x8*)(p);
  bf16x8 hi = *(const bf16x8*)(p + 16);
  bf16x16 o;
#pragma unroll
  for (int i = 0; i < 8; ++i) { o[i] = lo[i]; o[8 + i] = hi[i]; }
  return o;
}

// ---- B fragment (32x16 bf16) from an LDS tile of transposed weight rows:
//      row n holds K contiguous; lane = column n=lane%16, K half by lane/16.
//      STRIDE = rowBytes+16 pad -> bank stride 4 mod 64, 16B aligned. -------
template <int STRIDE>
__device__ __forceinline__ bf16x16
fragB_lds(const char* buf, int nLocal, int k0, int lane) {
  const int n  = nLocal + (lane & 15);
  const int kh = (lane & 16) ? 8 : 0;
  const char* p = buf + n * STRIDE + (k0 + kh) * 2;
  bf16x8 lo = *(const bf16x8*)(p);
  bf16x8 hi = *(const bf16x8*)(p + 32);
  bf16x16 o;
#pragma unroll
  for (int i = 0; i < 8; ++i) { o[i] = lo[i]; o[8 + i] = hi[i]; }
  return o;
}

__device__ __forceinline__ float sigmoidf_(float x) {
  return 1.0f / (1.0f + __expf(-x));
}

// ---------------------------------------------------------------------------
// Stage a 64-row x KT-col bf16 weight tile (rows nBase.., K window [kw,kw+KT))
// into LDS with row stride KT*2+16. TDM path: one TENSOR_LOAD_TO_LDS from
// wave 0 with pad_enable (pad 4 DWORDs every 2^(padCode+1) DWORDs == one row).
// ---------------------------------------------------------------------------
#if USE_TDM
__device__ __forceinline__ void
tdm_issue(const __bf16* gsrc, unsigned ldsAddr, int kElems, int ldkElems, int padCode) {
  unsigned long long ga = (unsigned long long)gsrc;
  u32x4 g0;
  g0[0] = 1u;                                    // count=1, user descriptor
  g0[1] = ldsAddr;                               // lds_addr
  g0[2] = (unsigned)(ga & 0xFFFFFFFFull);        // global_addr[31:0]
  g0[3] = (unsigned)((ga >> 32) & 0x1FFFFFFull)  // global_addr[56:32]
        | 0x80000000u;                           // type=2 ("image")
  i32x8 g1;
  g1[0] = (int)((1u << 16)                       // data_size = 2 bytes
              | (1u << 20)                       // pad_enable
              | ((unsigned)padCode << 22)        // pad_interval
              | (3u << 25));                     // pad_amount = 4 DWORDs (16B)
  g1[1] = (int)(((unsigned)kElems & 0xFFFFu) << 16);  // tensor_dim0 lo16
  g1[2] = (int)(64u << 16);                           // tensor_dim1 lo16
  g1[3] = (int)(((unsigned)kElems & 0xFFFFu) << 16);  // tile_dim0
  g1[4] = (int)64;                                    // tile_dim1 (tile_dim2=0)
  g1[5] = (int)(unsigned)ldkElems;                    // tensor_dim0_stride lo32
  g1[6] = 0;
  g1[7] = 0;
  i32x4 z4 = {0, 0, 0, 0};
#if TDM_ARITY == 6
  i32x8 z8 = {0, 0, 0, 0, 0, 0, 0, 0};
  __builtin_amdgcn_tensor_load_to_lds(g0, g1, z4, z4, z8, 0);
#else
  __builtin_amdgcn_tensor_load_to_lds(g0, g1, z4, z4, 0);
#endif
}
#endif

__device__ __forceinline__ void
stage_tile(const __bf16* __restrict__ wt, int ldk, int nBase, int kw,
           char* buf, int kElems, int padCode, int tid) {
  const __bf16* gsrc = wt + (long)nBase * ldk + kw;
#if USE_TDM
  if (tid < 32)                                   // wave 0 issues the DMA
    tdm_issue(gsrc, (unsigned)(unsigned long long)buf, kElems, ldk, padCode);
#else
  const int rowBytes = kElems * 2;
  const int chunks = rowBytes >> 4;               // 16B chunks per row
  for (int c = tid; c < 64 * chunks; c += (int)blockDim.x) {
    const int r = c / chunks, q = c - r * chunks;
    *(f32x4*)(buf + r * (rowBytes + 16) + q * 16) =
        *(const f32x4*)((const char*)(gsrc + (long)r * ldk) + q * 16);
  }
#endif
}

__device__ __forceinline__ void stage_commit(int tid) {
#if USE_TDM
  if (tid < 32) __builtin_amdgcn_s_wait_tensorcnt(0);
#else
  (void)tid;
#endif
  __syncthreads();
}

// ---- one 64xKT stage: acc[0..3] += A(16xKT strip) x Btile ------------------
// Statically typed A-source variants: no runtime pointer tests, and the
// straight-line use lets clang promote A loads to global_load_b128.
template <int KT, int LDA>
__device__ __forceinline__ void
gemm_stage_f32(f32x8 acc[4], const char* bufB, const float* __restrict__ a,
               int rowBase, int lane) {
#pragma unroll
  for (int k0 = 0; k0 < KT; k0 += 32) {
    bf16x16 af = fragA_f32<LDA>(a, rowBase, k0, lane);
#pragma unroll
    for (int t = 0; t < 4; ++t)
      acc[t] = wmma_bf16(af, fragB_lds<KT * 2 + 16>(bufB, t * 16, k0, lane), acc[t]);
  }
}

template <int KT, int LDA>
__device__ __forceinline__ void
gemm_stage_bf16(f32x8 acc[4], const char* bufB, const __bf16* __restrict__ a,
                int rowBase, int lane) {
#pragma unroll
  for (int k0 = 0; k0 < KT; k0 += 32) {
    bf16x16 af = fragA_bf16<LDA>(a, rowBase, k0, lane);
#pragma unroll
    for (int t = 0; t < 4; ++t)
      acc[t] = wmma_bf16(af, fragB_lds<KT * 2 + 16>(bufB, t * 16, k0, lane), acc[t]);
  }
}

// ---------------------------------------------------------------------------
// Kernel 0: weight transpose + f32->bf16  (W[K,N] row-major -> Wt[N,K] bf16)
// ---------------------------------------------------------------------------
__global__ void k_transpose(const float* __restrict__ W, __bf16* __restrict__ WT,
                            int K, int Nn) {
  int idx = blockIdx.x * blockDim.x + threadIdx.x;
  if (idx >= K * Nn) return;
  int n = idx / K;
  int k = idx - n * K;
  WT[idx] = (__bf16)W[(long)k * Nn + n];
}

// ---------------------------------------------------------------------------
// Kernel 1: x_proj = relu(x @ Wproj + b) -> bf16.  One 64x128 LDS stage.
// ---------------------------------------------------------------------------
__global__ __launch_bounds__(128) void
k_xproj(const float* __restrict__ x, const __bf16* __restrict__ WprojT,
        const float* __restrict__ bias, __bf16* __restrict__ XP) {
  __shared__ alignas(16) char sB[64 * (DNODE * 2 + 16)];     // 17 KB
  const int tid = threadIdx.x, lane = tid & 31, wave = tid >> 5;
  const int rowBase = blockIdx.x * 64 + wave * 16;
  const int nBase   = blockIdx.y * 64;
  stage_tile(WprojT, DNODE, nBase, 0, sB, DNODE, 5, tid);
  stage_commit(tid);
  f32x8 acc[4] = {};
  gemm_stage_f32<DNODE, DNODE>(acc, sB, x, rowBase, lane);
  const int mo = (lane & 16) ? 8 : 0;
#pragma unroll
  for (int t = 0; t < 4; ++t) {
    const int n = nBase + t * 16 + (lane & 15);
    const float bn = bias[n];
#pragma unroll
    for (int v = 0; v < 8; ++v) {
      float val = acc[t][v] + bn;
      val = val > 0.0f ? val : 0.0f;
      XP[(long)(rowBase + mo + v) * DPROJ + n] = (__bf16)val;
    }
  }
}

// ---------------------------------------------------------------------------
// Kernel 2: R = sigmoid([z|x_proj]@Wxr + h@Whr + b); store bf16(h*R).
// 3 double-buffered 64x256 stages: {Wxr|0..255, A=z} {Wxr|256.., A=XP}
// {Whr, A=h}.  TDM of stage s+1 overlaps WMMA of stage s.
// ---------------------------------------------------------------------------
__global__ __launch_bounds__(128) void
k_rgate(const float* __restrict__ z, const __bf16* __restrict__ XP,
        const float* __restrict__ ph,
        const __bf16* __restrict__ WxrT, const __bf16* __restrict__ WhrT,
        const float* __restrict__ bxr, const float* __restrict__ bhr,
        __bf16* __restrict__ HR) {
  __shared__ alignas(16) char sB[2][64 * (256 * 2 + 16)];    // 2 x 33 KB
  const int tid = threadIdx.x, lane = tid & 31, wave = tid >> 5;
  const int rowBase = blockIdx.x * 64 + wave * 16;
  const int nBase   = blockIdx.y * 64;

  f32x8 acc[4] = {};
  stage_tile(WxrT, DIN, nBase, 0, sB[0], 256, 6, tid);         // Wxr | k 0..255
  stage_commit(tid);

  stage_tile(WxrT, DIN, nBase, 256, sB[1], 256, 6, tid);       // Wxr | k 256..511
  gemm_stage_f32<256, DBB>(acc, sB[0], z, rowBase, lane);
  stage_commit(tid);

  stage_tile(WhrT, DBB, nBase, 0, sB[0], 256, 6, tid);         // Whr
  gemm_stage_bf16<256, DPROJ>(acc, sB[1], XP, rowBase, lane);
  stage_commit(tid);

  gemm_stage_f32<256, DBB>(acc, sB[0], ph, rowBase, lane);

  const int mo = (lane & 16) ? 8 : 0;
#pragma unroll
  for (int t = 0; t < 4; ++t) {
    const int n = nBase + t * 16 + (lane & 15);
    const float bn = bxr[n] + bhr[n];
#pragma unroll
    for (int v = 0; v < 8; ++v) {
      const int m = rowBase + mo + v;
      const float r = sigmoidf_(acc[t][v] + bn);
      HR[(long)m * DBB + n] = (__bf16)(ph[(long)m * DBB + n] * r);
    }
  }
}

// ---------------------------------------------------------------------------
// Kernel 3: Z and H~ fused, then GRU mix.  6 double-buffered 64x256 stages:
//   accZ: {Wxz|0,A=z} {Wxz|256,A=XP} {Whz,A=h}
//   accH: {Wxh|0,A=z} {Wxh|256,A=XP} {Whh,A=h*R}
// ---------------------------------------------------------------------------
__global__ __launch_bounds__(128) void
k_final(const float* __restrict__ z, const __bf16* __restrict__ XP,
        const float* __restrict__ ph, const __bf16* __restrict__ HR,
        const __bf16* __restrict__ WxzT, const __bf16* __restrict__ WhzT,
        const __bf16* __restrict__ WxhT, const __bf16* __restrict__ WhhT,
        const float* __restrict__ bxz, const float* __restrict__ bhz,
        const float* __restrict__ bxh, const float* __restrict__ bhh,
        float* __restrict__ hout) {
  __shared__ alignas(16) char sB[2][64 * (256 * 2 + 16)];    // 2 x 33 KB
  const int tid = threadIdx.x, lane = tid & 31, wave = tid >> 5;
  const int rowBase = blockIdx.x * 64 + wave * 16;
  const int nBase   = blockIdx.y * 64;

  f32x8 accZ[4] = {};
  f32x8 accH[4] = {};
  stage_tile(WxzT, DIN, nBase, 0, sB[0], 256, 6, tid);         // s0: Wxz | 0
  stage_commit(tid);

  stage_tile(WxzT, DIN, nBase, 256, sB[1], 256, 6, tid);       // s1: Wxz | 256
  gemm_stage_f32<256, DBB>(accZ, sB[0], z, rowBase, lane);
  stage_commit(tid);

  stage_tile(WhzT, DBB, nBase, 0, sB[0], 256, 6, tid);         // s2: Whz
  gemm_stage_bf16<256, DPROJ>(accZ, sB[1], XP, rowBase, lane);
  stage_commit(tid);

  stage_tile(WxhT, DIN, nBase, 0, sB[1], 256, 6, tid);         // s3: Wxh | 0
  gemm_stage_f32<256, DBB>(accZ, sB[0], ph, rowBase, lane);
  stage_commit(tid);

  stage_tile(WxhT, DIN, nBase, 256, sB[0], 256, 6, tid);       // s4: Wxh | 256
  gemm_stage_f32<256, DBB>(accH, sB[1], z, rowBase, lane);
  stage_commit(tid);

  stage_tile(WhhT, DBB, nBase, 0, sB[1], 256, 6, tid);         // s5: Whh
  gemm_stage_bf16<256, DPROJ>(accH, sB[0], XP, rowBase, lane);
  stage_commit(tid);

  gemm_stage_bf16<256, DBB>(accH, sB[1], HR, rowBase, lane);

  const int mo = (lane & 16) ? 8 : 0;
#pragma unroll
  for (int t = 0; t < 4; ++t) {
    const int n = nBase + t * 16 + (lane & 15);
    const float bz = bxz[n] + bhz[n];
    const float bh = bxh[n] + bhh[n];
#pragma unroll
    for (int v = 0; v < 8; ++v) {
      const int m = rowBase + mo + v;
      const float Zg = sigmoidf_(accZ[t][v] + bz);
      const float Ht = tanhf(accH[t][v] + bh);
      const float hp = ph[(long)m * DBB + n];
      __builtin_nontemporal_store(Zg * hp + (1.0f - Zg) * Ht,
                                  &hout[(long)m * DBB + n]);
    }
  }
}

// ---------------------------------------------------------------------------
// Kernel 4: mean readout  fused[g, 0:256] = mean_p z[g*22+p, :]
// ---------------------------------------------------------------------------
__global__ __launch_bounds__(256) void
k_readout(const float* __restrict__ z, float* __restrict__ fused) {
  const int g = blockIdx.x;
  const int d = threadIdx.x;
  float s = 0.0f;
#pragma unroll
  for (int p = 0; p < NPG; ++p)
    s += z[((long)g * NPG + p) * DBB + d];
  fused[(long)g * DFUSE + d] = s * (1.0f / (float)NPG);
}

// ---------------------------------------------------------------------------
// Kernel 5: glob linear  fused[g, 256:384] = relu(u @ Wglob + b)
// ---------------------------------------------------------------------------
__global__ __launch_bounds__(128) void
k_glob(const float* __restrict__ u, const __bf16* __restrict__ WglobT,
       const float* __restrict__ bias, float* __restrict__ fused) {
  __shared__ alignas(16) char sB[64 * (DGIN * 2 + 16)];      // 9 KB
  const int tid = threadIdx.x, lane = tid & 31, wave = tid >> 5;
  const int rowBase = blockIdx.x * 64 + wave * 16;
  const int nBase   = blockIdx.y * 64;
  stage_tile(WglobT, DGIN, nBase, 0, sB, DGIN, 4, tid);
  stage_commit(tid);
  f32x8 acc[4] = {};
  gemm_stage_f32<DGIN, DGIN>(acc, sB, u, rowBase, lane);
  const int mo = (lane & 16) ? 8 : 0;
#pragma unroll
  for (int t = 0; t < 4; ++t) {
    const int n = nBase + t * 16 + (lane & 15);
    const float bn = bias[n];
#pragma unroll
    for (int v = 0; v < 8; ++v) {
      float val = acc[t][v] + bn;
      val = val > 0.0f ? val : 0.0f;
      fused[(long)(rowBase + mo + v) * DFUSE + DBB + n] = val;
    }
  }
}

// ---------------------------------------------------------------------------
extern "C" void kernel_launch(void* const* d_in, const int* in_sizes, int n_in,
                              void* d_out, int out_size, void* d_ws, size_t ws_size,
                              hipStream_t stream) {
  (void)in_sizes; (void)n_in; (void)out_size; (void)ws_size;

  const float* z      = (const float*)d_in[0];
  const float* u      = (const float*)d_in[1];
  const float* x      = (const float*)d_in[2];
  // d_in[3] edge_index, d_in[4] batch, d_in[5] batch_size : unused (K=1 Cheb)
  const float* prev_h = (const float*)d_in[6];
  const float* w_proj = (const float*)d_in[7];
  const float* b_proj = (const float*)d_in[8];
  const float* w_glob = (const float*)d_in[9];
  const float* b_glob = (const float*)d_in[10];
  const float* W_xz = (const float*)d_in[11]; const float* b_xz = (const float*)d_in[12];
  const float* W_hz = (const float*)d_in[13]; const float* b_hz = (const float*)d_in[14];
  const float* W_xr = (const float*)d_in[15]; const float* b_xr = (const float*)d_in[16];
  const float* W_hr = (const float*)d_in[17]; const float* b_hr = (const float*)d_in[18];
  const float* W_xh = (const float*)d_in[19]; const float* b_xh = (const float*)d_in[20];
  const float* W_hh = (const float*)d_in[21]; const float* b_hh = (const float*)d_in[22];

  float* fused = (float*)d_out;                    // [4096, 384]
  float* hout  = fused + (size_t)BGR * DFUSE;      // [90112, 256]

  // ---- workspace layout (bf16) ----
  char* ws = (char*)d_ws;
  size_t o = 0;
  __bf16* WprojT = (__bf16*)(ws + o); o += (size_t)DPROJ * DNODE * 2;
  __bf16* WglobT = (__bf16*)(ws + o); o += (size_t)DGOUT * DGIN * 2;
  __bf16* WxzT   = (__bf16*)(ws + o); o += (size_t)DBB * DIN * 2;
  __bf16* WxrT   = (__bf16*)(ws + o); o += (size_t)DBB * DIN * 2;
  __bf16* WxhT   = (__bf16*)(ws + o); o += (size_t)DBB * DIN * 2;
  __bf16* WhzT   = (__bf16*)(ws + o); o += (size_t)DBB * DBB * 2;
  __bf16* WhrT   = (__bf16*)(ws + o); o += (size_t)DBB * DBB * 2;
  __bf16* WhhT   = (__bf16*)(ws + o); o += (size_t)DBB * DBB * 2;
  __bf16* XP     = (__bf16*)(ws + o); o += (size_t)NTOT * DPROJ * 2;   // 46 MB
  __bf16* HR     = (__bf16*)(ws + o); o += (size_t)NTOT * DBB * 2;     // 46 MB

  // ---- 0: weight transposes (f32 -> bf16, [K,N] -> [N,K]) ----
  auto tgrid = [](int K, int Nn) { return dim3((unsigned)((K * Nn + 255) / 256)); };
  k_transpose<<<tgrid(DNODE, DPROJ), 256, 0, stream>>>(w_proj, WprojT, DNODE, DPROJ);
  k_transpose<<<tgrid(DGIN,  DGOUT), 256, 0, stream>>>(w_glob, WglobT, DGIN,  DGOUT);
  k_transpose<<<tgrid(DIN,   DBB),   256, 0, stream>>>(W_xz,   WxzT,   DIN,   DBB);
  k_transpose<<<tgrid(DIN,   DBB),   256, 0, stream>>>(W_xr,   WxrT,   DIN,   DBB);
  k_transpose<<<tgrid(DIN,   DBB),   256, 0, stream>>>(W_xh,   WxhT,   DIN,   DBB);
  k_transpose<<<tgrid(DBB,   DBB),   256, 0, stream>>>(W_hz,   WhzT,   DBB,   DBB);
  k_transpose<<<tgrid(DBB,   DBB),   256, 0, stream>>>(W_hr,   WhrT,   DBB,   DBB);
  k_transpose<<<tgrid(DBB,   DBB),   256, 0, stream>>>(W_hh,   WhhT,   DBB,   DBB);

  // ---- fused output (independent of GRU path) ----
  k_readout<<<dim3(BGR), 256, 0, stream>>>(z, fused);
  k_glob<<<dim3(BGR / 64, DGOUT / 64), 128, 0, stream>>>(u, WglobT, b_glob, fused);

  // ---- GRU path ----
  const dim3 ggrid(NTOT / 64, DBB / 64);           // 1408 x 4, 4 waves/block
  k_xproj<<<ggrid, 128, 0, stream>>>(x, WprojT, b_proj, XP);
  k_rgate<<<ggrid, 128, 0, stream>>>(z, XP, prev_h, WxrT, WhrT, b_xr, b_hr, HR);
  k_final<<<ggrid, 128, 0, stream>>>(z, XP, prev_h, HR,
                                     WxzT, WhzT, WxhT, WhhT,
                                     b_xz, b_hz, b_xh, b_hh, hout);
}
